// GraphRegressor_18889266167943
// MI455X (gfx1250) — matrix operand order repaired and checked by reading, not compiled, not run on hardware
//
#include <hip/hip_runtime.h>

typedef __bf16 bf16_t;
typedef __attribute__((ext_vector_type(16))) __bf16 v16bf;
typedef __attribute__((ext_vector_type(8)))  float  v8f;
typedef __attribute__((ext_vector_type(4)))  unsigned int v4u;
typedef __attribute__((ext_vector_type(8)))  int v8i;
typedef __attribute__((ext_vector_type(4)))  int v4i;

#define Bsz 128
#define Nn  1024
#define Hd  128
#define NQ  512
#define NX  256
#define EPS 1e-5f

__device__ __forceinline__ bf16_t f2bf(float f) { return (bf16_t)f; }

template <typename T>
__device__ __forceinline__ unsigned lds_off(T* p) {
  return (unsigned)(unsigned long long)(void*)p;   // low 32 bits = LDS byte offset
}

// Build a 2-D TDM descriptor and issue TENSOR_LOAD_TO_LDS (clang-23 6-arg form).
// data_size = 2 bytes (bf16). Tile = tile_d0 (contig) x tile_d1 rows.
__device__ __forceinline__ void tdm_load_2d(unsigned ldsoff, unsigned long long gaddr,
                                            unsigned tile_d0, unsigned tile_d1,
                                            unsigned tens_d0, unsigned tens_d1,
                                            unsigned long long stride0) {
  v4u g0;
  g0[0] = 1u;                                               // count=1, user mode
  g0[1] = ldsoff;                                           // lds_addr
  g0[2] = (unsigned)(gaddr & 0xFFFFFFFFull);                // global_addr lo
  g0[3] = (unsigned)((gaddr >> 32) & 0x1FFFFFFull) | (2u << 30);  // hi + type=2
  v8i g1;
  g1[0] = (int)(1u << 16);                                  // data_size = 2B
  g1[1] = (int)((tens_d0 & 0xFFFFu) << 16);                 // tensor_dim0[15:0]
  g1[2] = (int)(((tens_d0 >> 16) & 0xFFFFu) | ((tens_d1 & 0xFFFFu) << 16));
  g1[3] = (int)(((tens_d1 >> 16) & 0xFFFFu) | ((tile_d0 & 0xFFFFu) << 16));
  g1[4] = (int)(tile_d1 & 0xFFFFu);                         // tile_dim1, tile_dim2=0
  g1[5] = (int)(unsigned)(stride0 & 0xFFFFFFFFull);         // dim0 stride lo
  g1[6] = (int)(unsigned)((stride0 >> 32) & 0xFFFFull);     // dim0 stride hi
  g1[7] = 0;
  v4i z4 = {0, 0, 0, 0};
  v8i z8 = {0, 0, 0, 0, 0, 0, 0, 0};
  __builtin_amdgcn_tensor_load_to_lds(g0, g1, z4, z4, z8, 0);
}

// ---------------------------------------------------------------------------
// Prep: adj int32 [b][j][i] -> ATbf bf16 [b][i][j]  (LDS 64x64 tile transpose)
// fused with layer-0 segment degree sums (deg[b,i,{q,x,z}]).
// ---------------------------------------------------------------------------
__global__ __launch_bounds__(256) void k_prep(const int* __restrict__ adj,
                                              bf16_t* __restrict__ ATbf,
                                              float* __restrict__ deg) {
  __shared__ bf16_t lT[64 * 64];      // lT[il*64 + jl]
  __shared__ float  lsum[256];
  int b = blockIdx.z, j0 = blockIdx.y * 64, i0 = blockIdx.x * 64;
  int tid = threadIdx.x;
  const int* adjb = adj + (size_t)b * Nn * Nn;
#pragma unroll
  for (int r = 0; r < 16; ++r) {                 // coalesced over i
    int idx = r * 256 + tid;
    int il = idx & 63, jl = idx >> 6;
    lT[il * 64 + jl] = f2bf((float)adjb[(size_t)(j0 + jl) * Nn + i0 + il]);
  }
  __syncthreads();
  // transposed write-out, coalesced over j (uint4 = 8 bf16)
#pragma unroll
  for (int r = 0; r < 2; ++r) {
    int idx = r * 256 + tid;
    int jg = idx & 7, il = idx >> 3;
    *(uint4*)&ATbf[((size_t)b * Nn + i0 + il) * Nn + j0 + jg * 8] =
        *(const uint4*)&lT[il * 64 + jg * 8];
  }
  // degree partials: 4 threads per i, each sums 16 j
  {
    int il = tid & 63, q = tid >> 6;
    float s = 0.f;
#pragma unroll
    for (int jj = 0; jj < 16; ++jj) s += (float)lT[il * 64 + q * 16 + jj];
    lsum[tid] = s;
  }
  __syncthreads();
  if (tid < 64) {
    float tot = lsum[tid] + lsum[tid + 64] + lsum[tid + 128] + lsum[tid + 192];
    int seg = (j0 >= NQ + NX) ? 2 : ((j0 >= NQ) ? 1 : 0);
    atomicAdd(&deg[((size_t)b * Nn + i0 + tid) * 3 + seg], tot);
  }
}

// ---------------------------------------------------------------------------
// agg0[b,i,c] = x[i,c] + deg_q*xq[c] + deg_x*xx[c] + deg_z*xz[c]
// Input-BN stats of the fixed one-hot features are analytic constants.
// ---------------------------------------------------------------------------
__global__ __launch_bounds__(256) void k_agg0(const float* __restrict__ deg,
                                              const float* __restrict__ big,
                                              const float* __restrict__ bib,
                                              float* __restrict__ agg0) {
  size_t idx = (size_t)blockIdx.x * 256 + threadIdx.x;   // b*N + i
  int i = (int)(idx & (Nn - 1));
  const float mean_[3] = {0.5f, 0.25f, 0.25f};
  const float var_[3]  = {0.25f, 0.1875f, 0.1875f};
  float xv[3][3];
#pragma unroll
  for (int c = 0; c < 3; ++c) {
    float rs = rsqrtf(var_[c] + EPS) * big[c];
    xv[0][c] = ((c == 0 ? 1.f : 0.f) - mean_[c]) * rs + bib[c];
    xv[1][c] = ((c == 1 ? 1.f : 0.f) - mean_[c]) * rs + bib[c];
    xv[2][c] = ((c == 2 ? 1.f : 0.f) - mean_[c]) * rs + bib[c];
  }
  float dq = deg[idx * 3], dx = deg[idx * 3 + 1], dz = deg[idx * 3 + 2];
  int ti = (i < NQ) ? 0 : (i < NQ + NX ? 1 : 2);
#pragma unroll
  for (int c = 0; c < 3; ++c)
    agg0[idx * 3 + c] = xv[ti][c] + dq * xv[0][c] + dx * xv[1][c] + dz * xv[2][c];
}

// Layer-0 MLP part A: t = relu(agg0 @ w0a + b0a), K=3 -> VALU.
__global__ __launch_bounds__(256) void k_mlp0a(const float* __restrict__ agg0,
                                               const float* __restrict__ w,
                                               const float* __restrict__ bias,
                                               bf16_t* __restrict__ out) {
  size_t idx = (size_t)blockIdx.x * 256 + threadIdx.x;
  size_t r = idx >> 7;
  int n = (int)(idx & 127);
  float v = agg0[r * 3] * w[n] + agg0[r * 3 + 1] * w[Hd + n] +
            agg0[r * 3 + 2] * w[2 * Hd + n] + bias[n];
  out[idx] = f2bf(v > 0.f ? v : 0.f);
}

// Weight prep: W[k][n] f32 -> WT[n][k] bf16 (done once; reused by 1024 blocks).
__global__ __launch_bounds__(256) void k_prepw(const float* __restrict__ W,
                                               bf16_t* __restrict__ WT) {
  int idx = blockIdx.x * 256 + threadIdx.x;
  int n = idx >> 7, k = idx & 127;
  WT[n * 128 + k] = f2bf(W[k * 128 + n]);
}

// ---------------------------------------------------------------------------
// WMMA aggregation: aggbf[b,i,d] = h[b,i,d] + sum_j AT[b,i,j]*hT[b,d,j]
// Both operand tiles (128x32, contiguous rows) loaded via TDM, double-buffered
// on TENSORcnt. 8 waves: wave w -> i-rows [w*16,w*16+16) x all 128 d.
// ---------------------------------------------------------------------------
__global__ __launch_bounds__(256) void k_agg(const bf16_t* __restrict__ hbfT,
                                             const float* __restrict__ hf,
                                             const bf16_t* __restrict__ ATbf,
                                             bf16_t* __restrict__ aggbf) {
  __shared__ bf16_t lA[2][128 * 32];
  __shared__ bf16_t lH[2][128 * 32];
  int b = blockIdx.y, i0 = blockIdx.x * 128;
  int tid = threadIdx.x;
  int wave = tid >> 5, lane = tid & 31, lrow = lane & 15, lhi = lane >> 4;
  v8f acc[8] = {};

  unsigned long long baseA = (unsigned long long)(const void*)ATbf +
                             ((size_t)b * Nn + i0) * Nn * 2ull;
  unsigned long long baseH = (unsigned long long)(const void*)hbfT +
                             (size_t)b * Hd * Nn * 2ull;
  if (tid < 32) {
    tdm_load_2d(lds_off(&lA[0][0]), baseA, 32, 128, Nn, Nn, Nn);
    tdm_load_2d(lds_off(&lH[0][0]), baseH, 32, 128, Nn, Hd, Nn);
  }
  for (int kc = 0; kc < Nn / 32; ++kc) {
    int cur = kc & 1;
    if (tid < 32) {
      if (kc < Nn / 32 - 1) {
        unsigned long long off = (unsigned long long)(kc + 1) * 32 * 2;
        tdm_load_2d(lds_off(&lA[cur ^ 1][0]), baseA + off, 32, 128, Nn, Nn, Nn);
        tdm_load_2d(lds_off(&lH[cur ^ 1][0]), baseH + off, 32, 128, Nn, Hd, Nn);
        __builtin_amdgcn_s_wait_tensorcnt(2);   // current pair complete
      } else {
        __builtin_amdgcn_s_wait_tensorcnt(0);
      }
    }
    __syncthreads();
    const bf16_t* pa = &lA[cur][(wave * 16 + lrow) * 32 + (lhi ? 16 : 0)];
    v16bf af;
#pragma unroll
    for (int e = 0; e < 16; ++e) af[e] = pa[e];
#pragma unroll
    for (int t = 0; t < 8; ++t) {
      const bf16_t* pb = &lH[cur][(t * 16 + lrow) * 32 + (lhi ? 16 : 0)];
      v16bf bf;
#pragma unroll
      for (int e = 0; e < 16; ++e) bf[e] = pb[e];
      acc[t] = __builtin_amdgcn_wmma_f32_16x16x32_bf16(
          false, af, false, bf, (short)0, acc[t], false, false);
    }
    __syncthreads();   // all waves done before buffer is overwritten
  }
  int ibase = i0 + wave * 16 + (lhi ? 8 : 0);
#pragma unroll
  for (int t = 0; t < 8; ++t) {
    int d = t * 16 + lrow;
#pragma unroll
    for (int v = 0; v < 8; ++v) {
      size_t off = ((size_t)b * Nn + (ibase + v)) * Hd + d;
      aggbf[off] = f2bf(acc[t][v] + hf[off]);   // + self term (eps=0 GIN)
    }
  }
}

// ---------------------------------------------------------------------------
// WMMA GEMM: [R,128]bf16 @ WT(bf16,[n][k]) + bias.  X tiles TDM double-buffered.
// MODE 1: leaky->bf16, MODE 2: linear->f32.
// ---------------------------------------------------------------------------
template <int MODE>
__global__ __launch_bounds__(256) void k_gemm(const bf16_t* __restrict__ X,
                                              const bf16_t* __restrict__ WT,
                                              const float* __restrict__ bias,
                                              bf16_t* __restrict__ outb,
                                              float* __restrict__ outf) {
  __shared__ __align__(16) bf16_t lW[128 * 128];   // [n][k]
  __shared__ bf16_t lX[2][128 * 32];
  int tid = threadIdx.x;
  size_t r0 = (size_t)blockIdx.x * 128;
  {
    const uint4* src = (const uint4*)WT;
    uint4* dst = (uint4*)lW;
#pragma unroll
    for (int r = 0; r < 8; ++r) dst[r * 256 + tid] = src[r * 256 + tid];
  }
  unsigned long long baseX = (unsigned long long)(const void*)X + r0 * Hd * 2ull;
  if (tid < 32) tdm_load_2d(lds_off(&lX[0][0]), baseX, 32, 128, Hd, 128, Hd);

  int wave = tid >> 5, lane = tid & 31, lrow = lane & 15, lhi = lane >> 4;
  v8f acc[8] = {};
  for (int kc = 0; kc < 4; ++kc) {
    int cur = kc & 1;
    if (tid < 32) {
      if (kc < 3) {
        tdm_load_2d(lds_off(&lX[cur ^ 1][0]),
                    baseX + (unsigned long long)(kc + 1) * 32 * 2, 32, 128, Hd, 128, Hd);
        __builtin_amdgcn_s_wait_tensorcnt(1);
      } else {
        __builtin_amdgcn_s_wait_tensorcnt(0);
      }
    }
    __syncthreads();
    int k0 = kc * 32;
    const bf16_t* pa = &lX[cur][(wave * 16 + lrow) * 32 + (lhi ? 16 : 0)];
    v16bf af;
#pragma unroll
    for (int e = 0; e < 16; ++e) af[e] = pa[e];
#pragma unroll
    for (int t = 0; t < 8; ++t) {
      const bf16_t* pb = &lW[(t * 16 + lrow) * 128 + k0 + (lhi ? 16 : 0)];
      v16bf bf;
#pragma unroll
      for (int e = 0; e < 16; ++e) bf[e] = pb[e];
      acc[t] = __builtin_amdgcn_wmma_f32_16x16x32_bf16(
          false, af, false, bf, (short)0, acc[t], false, false);
    }
    __syncthreads();
  }
  size_t rbase = r0 + wave * 16 + (lhi ? 8 : 0);
#pragma unroll
  for (int t = 0; t < 8; ++t) {
    int n = t * 16 + lrow;
    float bia = bias[n];
#pragma unroll
    for (int v = 0; v < 8; ++v) {
      float val = acc[t][v] + bia;
      size_t off = (rbase + v) * Hd + n;
      if (MODE == 1) outb[off] = f2bf(val > 0.f ? val : 0.1f * val);
      else           outf[off] = val;
    }
  }
}

// --------------------------- BatchNorm machinery ---------------------------
__global__ void k_zero(float* p, int n) {
  int i = blockIdx.x * 256 + threadIdx.x;
  if (i < n) p[i] = 0.f;
}

__global__ __launch_bounds__(256) void k_stats(const float* __restrict__ m,
                                               float* __restrict__ stats) {
  __shared__ float ss[256], sq[256];
  int tid = threadIdx.x;
  int n = tid & 127, half = tid >> 7;
  size_t r0 = (size_t)blockIdx.x * 128;
  float s = 0.f, q = 0.f;
  for (int rr = half; rr < 128; rr += 2) {
    float v = m[(r0 + rr) * Hd + n];
    s += v; q += v * v;
  }
  ss[tid] = s; sq[tid] = q;
  __syncthreads();
  if (tid < 128) {
    atomicAdd(&stats[n],       ss[tid] + ss[tid + 128]);
    atomicAdd(&stats[128 + n], sq[tid] + sq[tid + 128]);
  }
}

// BN + residual + relu -> h (f32, [b][i][n]) and transposed bf16 hbfT[b][n][i]
__global__ __launch_bounds__(256) void k_bnapply(const float* __restrict__ m,
                                                 const float* __restrict__ stats,
                                                 const float* __restrict__ g,
                                                 const float* __restrict__ bb,
                                                 int res,
                                                 float* __restrict__ h,
                                                 bf16_t* __restrict__ hbfT) {
  __shared__ __align__(16) bf16_t lT[128 * 128];   // [n][il]
  int b = blockIdx.y, i0 = blockIdx.x * 128;
  int tid = threadIdx.x;
  const float inv_cnt = 1.f / ((float)Bsz * (float)Nn);
#pragma unroll 4
  for (int r = 0; r < 64; ++r) {
    int idx = r * 256 + tid;
    int n = idx & 127, il = idx >> 7;
    size_t off = ((size_t)b * Nn + i0 + il) * Hd + n;
    float mu = stats[n] * inv_cnt;
    float var = stats[128 + n] * inv_cnt - mu * mu;
    float v = (m[off] - mu) * rsqrtf(var + EPS) * g[n] + bb[n];
    if (res) v += h[off];
    v = v > 0.f ? v : 0.f;
    h[off] = v;
    lT[n * 128 + il] = f2bf(v);
  }
  __syncthreads();
#pragma unroll
  for (int r = 0; r < 8; ++r) {                    // transposed bf16 out, uint4
    int idx = r * 256 + tid;
    int ig = idx & 15, n = idx >> 4;
    *(uint4*)&hbfT[((size_t)b * Hd + n) * Nn + i0 + ig * 8] =
        *(const uint4*)&lT[n * 128 + ig * 8];
  }
}

// ----------------------------- Head kernels --------------------------------
__global__ __launch_bounds__(128) void k_pool_ln(const float* __restrict__ h,
                                                 const float* __restrict__ g,
                                                 const float* __restrict__ bb,
                                                 float* __restrict__ out) {
  int b = blockIdx.x, d = threadIdx.x;
  const float* hb = h + (size_t)b * Nn * Hd;
  float s = 0.f;
  for (int i = 0; i < Nn; ++i) s += hb[(size_t)i * Hd + d];
  float p = s * (1.f / Nn);
  __shared__ float red[128];
  red[d] = p;
  for (int o = 64; o > 0; o >>= 1) { __syncthreads(); if (d < o) red[d] += red[d + o]; }
  __syncthreads();
  float mu = red[0] * (1.f / Hd);
  __syncthreads();
  red[d] = (p - mu) * (p - mu);
  for (int o = 64; o > 0; o >>= 1) { __syncthreads(); if (d < o) red[d] += red[d + o]; }
  __syncthreads();
  float var = red[0] * (1.f / Hd);
  out[b * Hd + d] = (p - mu) * rsqrtf(var + EPS) * g[d] + bb[d];
}

__global__ void k_dense(const float* __restrict__ X, const float* __restrict__ W,
                        const float* __restrict__ bias, float* __restrict__ out,
                        int IN, int OUT) {
  int idx = blockIdx.x * 256 + threadIdx.x;
  int r = idx / OUT, n = idx % OUT;
  float s = bias[n];
  for (int k = 0; k < IN; ++k) s += X[r * IN + k] * W[k * OUT + n];
  out[idx] = s;
}

__global__ __launch_bounds__(128) void k_fcbn(float* __restrict__ z,
                                              const float* __restrict__ g,
                                              const float* __restrict__ bb,
                                              int OUT) {
  int n = blockIdx.x, b = threadIdx.x;
  float v = z[b * OUT + n];
  __shared__ float red[128];
  red[b] = v;
  for (int o = 64; o > 0; o >>= 1) { __syncthreads(); if (b < o) red[b] += red[b + o]; }
  __syncthreads();
  float mu = red[0] * (1.f / 128.f);
  __syncthreads();
  red[b] = (v - mu) * (v - mu);
  for (int o = 64; o > 0; o >>= 1) { __syncthreads(); if (b < o) red[b] += red[b + o]; }
  __syncthreads();
  float var = red[0] * (1.f / 128.f);
  float y = (v - mu) * rsqrtf(var + EPS) * g[n] + bb[n];
  z[b * OUT + n] = y > 0.f ? y : 0.1f * y;
}

__global__ __launch_bounds__(128) void k_fc3(const float* __restrict__ z,
                                             const float* __restrict__ w,
                                             const float* __restrict__ bias,
                                             float* __restrict__ out) {
  int b = threadIdx.x;
  float s = bias[0];
  for (int k = 0; k < 512; ++k) s += z[b * 512 + k] * w[k];
  out[b] = s;
}

// ---------------------------------------------------------------------------
extern "C" void kernel_launch(void* const* d_in, const int* in_sizes, int n_in,
                              void* d_out, int out_size, void* d_ws, size_t ws_size,
                              hipStream_t stream) {
  (void)in_sizes; (void)n_in; (void)out_size; (void)ws_size;
  const int*   adj   = (const int*)d_in[0];
  const float* wa[3] = {(const float*)d_in[1], (const float*)d_in[5], (const float*)d_in[9]};
  const float* ba[3] = {(const float*)d_in[2], (const float*)d_in[6], (const float*)d_in[10]};
  const float* wb[3] = {(const float*)d_in[3], (const float*)d_in[7], (const float*)d_in[11]};
  const float* bbv[3]= {(const float*)d_in[4], (const float*)d_in[8], (const float*)d_in[12]};
  const float* bn_in_g = (const float*)d_in[13];
  const float* bn_in_b = (const float*)d_in[14];
  const float* bng[3] = {(const float*)d_in[15], (const float*)d_in[17], (const float*)d_in[19]};
  const float* bnb[3] = {(const float*)d_in[16], (const float*)d_in[18], (const float*)d_in[20]};
  const float* ln_g = (const float*)d_in[21];
  const float* ln_b = (const float*)d_in[22];
  const float* fc1_w = (const float*)d_in[23]; const float* fc1_b = (const float*)d_in[24];
  const float* fcbn1_g = (const float*)d_in[25]; const float* fcbn1_b = (const float*)d_in[26];
  const float* fc2_w = (const float*)d_in[27]; const float* fc2_b = (const float*)d_in[28];
  const float* fcbn2_g = (const float*)d_in[29]; const float* fcbn2_b = (const float*)d_in[30];
  const float* fc3_w = (const float*)d_in[31]; const float* fc3_b = (const float*)d_in[32];

  char* ws = (char*)d_ws;
  const size_t RH  = (size_t)Bsz * Nn * Hd;          // 16,777,216
  const size_t BNN = (size_t)Bsz * Nn * Nn;          // 134,217,728
  float*  h     = (float*) (ws);                      // 64 MB
  float*  m     = (float*) (ws + RH * 4);             // 64 MB
  bf16_t* hbfT  = (bf16_t*)(ws + RH * 8);             // 32 MB  [b][d][i]
  bf16_t* aggbf = (bf16_t*)(ws + RH * 8 + RH * 2);    // 32 MB
  bf16_t* tbf   = (bf16_t*)(ws + RH * 8 + RH * 4);    // 32 MB
  bf16_t* ATbf  = (bf16_t*)(ws + RH * 8 + RH * 6);    // 256 MB [b][i][j]
  char*   tail  = ws + RH * 8 + RH * 6 + BNN * 2;
  float*  deg   = (float*)tail;                       // B*N*3
  float*  agg0  = deg + (size_t)Bsz * Nn * 3;
  bf16_t* wt[5];
  {
    bf16_t* wbase = (bf16_t*)(agg0 + (size_t)Bsz * Nn * 3);
    for (int i = 0; i < 5; ++i) wt[i] = wbase + (size_t)i * 128 * 128;
  }
  float* stats = (float*)(wt[4] + 128 * 128);
  float* lnout = stats + 1024;
  float* z1    = lnout + Bsz * Hd;
  float* z2    = z1 + Bsz * 256;

  const int RB = Bsz * Nn / 128;                      // 1024

  // --- prep: A^T bf16 + degrees, weight conversions ---
  k_zero <<<(Bsz * Nn * 3 + 255) / 256, 256, 0, stream>>>(deg, Bsz * Nn * 3);
  k_prep <<<dim3(Nn / 64, Nn / 64, Bsz), 256, 0, stream>>>(adj, ATbf, deg);
  k_agg0 <<<Bsz * Nn / 256, 256, 0, stream>>>(deg, bn_in_g, bn_in_b, agg0);
  k_prepw<<<64, 256, 0, stream>>>(wb[0], wt[0]);
  k_prepw<<<64, 256, 0, stream>>>(wa[1], wt[1]);
  k_prepw<<<64, 256, 0, stream>>>(wb[1], wt[2]);
  k_prepw<<<64, 256, 0, stream>>>(wa[2], wt[3]);
  k_prepw<<<64, 256, 0, stream>>>(wb[2], wt[4]);

  // --- 3 GIN layers ---
  for (int l = 0; l < 3; ++l) {
    if (l == 0) {
      k_mlp0a<<<(int)(RH / 256), 256, 0, stream>>>(agg0, wa[0], ba[0], tbf);
    } else {
      k_agg<<<dim3(Nn / 128, Bsz), 256, 0, stream>>>(hbfT, h, ATbf, aggbf);
      k_gemm<1><<<RB, 256, 0, stream>>>(aggbf, wt[2 * l - 1], ba[l], tbf, nullptr);
    }
    k_gemm<2><<<RB, 256, 0, stream>>>(tbf, wt[2 * l], bbv[l], nullptr, m);
    k_zero  <<<1, 256, 0, stream>>>(stats, 256);
    k_stats <<<RB, 256, 0, stream>>>(m, stats);
    k_bnapply<<<dim3(Nn / 128, Bsz), 256, 0, stream>>>(m, stats, bng[l], bnb[l],
                                                       (l > 0) ? 1 : 0, h, hbfT);
  }

  // --- head ---
  k_pool_ln<<<Bsz, 128, 0, stream>>>(h, ln_g, ln_b, lnout);
  k_dense  <<<Bsz * 256 / 256, 256, 0, stream>>>(lnout, fc1_w, fc1_b, z1, 128, 256);
  k_fcbn   <<<256, 128, 0, stream>>>(z1, fcbn1_g, fcbn1_b, 256);
  k_dense  <<<Bsz * 512 / 256, 256, 0, stream>>>(z1, fc2_w, fc2_b, z2, 256, 512);
  k_fcbn   <<<512, 128, 0, stream>>>(z2, fcbn2_g, fcbn2_b, 512);
  k_fc3    <<<1, 128, 0, stream>>>(z2, fc3_w, fc3_b, (float*)d_out);
}
